// GAT_75265006895626
// MI455X (gfx1250) — matrix operand order repaired and checked by reading, not compiled, not run on hardware
//
#include <hip/hip_runtime.h>

// ---------------------------------------------------------------------------
// GAT forward for MI455X (gfx1250, wave32, WMMA).
// Pipeline:
//   0) pack adj -> bitmask (ballot_w32), cast x/Ws/Wo to bf16 (weights transposed)
//   1) Wh  = x @ Wcat            (bf16 WMMA GEMM, f32 acc)  -> V  [4096x512] bf16
//   2) s1/s2 per head, s2 row-max bound
//   3) fused masked-softmax attention + P@V per head (WMMA), ELU -> hcat bf16
//   4) Whc = hcat @ Wo           (same GEMM)                -> V  (reused)
//   5) fused attention over full 512 width, ELU -> d_out f32
// ---------------------------------------------------------------------------

typedef __bf16  v16bf __attribute__((ext_vector_type(16)));
typedef __bf16  v8bf  __attribute__((ext_vector_type(8)));
typedef float   v8f   __attribute__((ext_vector_type(8)));
typedef float   v4f   __attribute__((ext_vector_type(4)));

#define NN    4096
#define FD    512
#define HEADS 8
#define NHD   64
#define ALPHA 0.2f
#define NEGV  -9.0e15f

__device__ inline v16bf pack16(v8bf a, v8bf b) {
    v16bf r;
#pragma unroll
    for (int e = 0; e < 8; ++e) { r[e] = a[e]; r[8 + e] = b[e]; }
    return r;
}

// ---------------- prep kernels --------------------------------------------

__global__ void k_cvt_x(const float* __restrict__ x, __bf16* __restrict__ xb) {
    int idx = blockIdx.x * blockDim.x + threadIdx.x;   // NN*FD threads
    xb[idx] = (__bf16)x[idx];
}

// WcatT[c][f] = Ws[h][f][k]  (c = h*64+k) ; WoT[c][f] = Wo[f][c]
__global__ void k_prep_w(const float* __restrict__ Ws, const float* __restrict__ Wo,
                         __bf16* __restrict__ WcatT, __bf16* __restrict__ WoT) {
    int idx = blockIdx.x * blockDim.x + threadIdx.x;   // FD*FD threads
    int c = idx / FD, f = idx % FD;
    int h = c >> 6, k = c & 63;
    WcatT[c * FD + f] = (__bf16)Ws[(h * FD + f) * NHD + k];
    WoT[c * FD + f]   = (__bf16)Wo[f * FD + c];
}

__global__ void k_pack_adj(const float* __restrict__ adj, unsigned* __restrict__ mask) {
    int idx = blockIdx.x * blockDim.x + threadIdx.x;   // NN*NN threads
    unsigned b = __builtin_amdgcn_ballot_w32(adj[idx] > 0.0f);
    if ((threadIdx.x & 31) == 0) mask[idx >> 5] = b;
}

// tiled bf16 transpose: Vt[c][r] = V[r][c], V is [NN][FD]
__global__ void k_transpose(const __bf16* __restrict__ V, __bf16* __restrict__ Vt) {
    __shared__ __bf16 tile[32][33];
    int tx = threadIdx.x, ty = threadIdx.y;            // block (32,8)
    int c = blockIdx.x * 32 + tx;                      // col in V
#pragma unroll
    for (int k = 0; k < 4; ++k) {
        int r = blockIdx.y * 32 + ty + k * 8;
        tile[ty + k * 8][tx] = V[r * FD + c];
    }
    __syncthreads();
    int ro = blockIdx.y * 32 + tx;                     // row in V -> col in Vt
#pragma unroll
    for (int k = 0; k < 4; ++k) {
        int co = blockIdx.x * 32 + ty + k * 8;         // col in V -> row in Vt
        Vt[co * NN + ro] = tile[tx][ty + k * 8];
    }
}

// ---------------- GEMM: C[M=NN][FD] = A[NN][FD] @ B, B given transposed -----

__global__ __launch_bounds__(256)
void k_gemm(const __bf16* __restrict__ A, const __bf16* __restrict__ BT,
            __bf16* __restrict__ C) {
    int wave = threadIdx.x >> 5, lane = threadIdx.x & 31;
    int r0 = blockIdx.x * 128 + wave * 16;
    int c0 = blockIdx.y * 64;
    int ln = lane & 15;
    int hiA = (lane < 16) ? 0 : 16;                    // A operand K halves
    int hiB = (lane < 16) ? 0 : 16;                    // B operand K halves
    int arow = r0 + ln;

    v8f acc[4];
#pragma unroll
    for (int t = 0; t < 4; ++t)
#pragma unroll
        for (int e = 0; e < 8; ++e) acc[t][e] = 0.0f;

    for (int ks = 0; ks < FD; ks += 32) {
        // issue all operand loads for this K-step up front; one wait covers
        // the 4 WMMAs below.
        v8bf a0 = *(const v8bf*)(A + arow * FD + ks + hiA);
        v8bf a1 = *(const v8bf*)(A + arow * FD + ks + hiA + 8);
        v16bf bv[4];
#pragma unroll
        for (int nt = 0; nt < 4; ++nt) {
            int bcol = c0 + nt * 16 + ln;
            bv[nt] = *(const v16bf*)(BT + bcol * FD + ks + hiB);
        }
        v16bf av = pack16(a0, a1);
#pragma unroll
        for (int nt = 0; nt < 4; ++nt) {
            acc[nt] = __builtin_amdgcn_wmma_f32_16x16x32_bf16(
                false, av, false, bv[nt], (short)0, acc[nt], false, false);
        }
    }
#pragma unroll
    for (int nt = 0; nt < 4; ++nt) {
        int col = c0 + nt * 16 + ln;
#pragma unroll
        for (int v = 0; v < 8; ++v) {
            int row = r0 + v + ((lane < 16) ? 0 : 8);
            C[row * FD + col] = (__bf16)acc[nt][v];
        }
    }
}

// ---------------- scores: s1/s2 = C(head slice) . a --------------------------

__global__ void k_scores(const __bf16* __restrict__ C, const float* __restrict__ a1,
                         const float* __restrict__ a2, int nheads, int nh,
                         float* __restrict__ s1, float* __restrict__ s2) {
    int idx = blockIdx.x * blockDim.x + threadIdx.x;   // nheads*NN threads
    int h = idx / NN, i = idx % NN;
    const __bf16* row = C + i * FD + h * nh;
    float r1 = 0.0f, r2 = 0.0f;
    for (int k = 0; k < nh; ++k) {
        float v = (float)row[k];
        r1 += v * a1[h * nh + k];
        r2 += v * a2[h * nh + k];
    }
    s1[h * NN + i] = r1;
    s2[h * NN + i] = r2;
}

__global__ void k_smax(const float* __restrict__ s2, float* __restrict__ smax) {
    int h = blockIdx.x;
    float m = -3.0e38f;
    for (int j = threadIdx.x; j < NN; j += blockDim.x) m = fmaxf(m, s2[h * NN + j]);
#pragma unroll
    for (int off = 16; off > 0; off >>= 1) m = fmaxf(m, __shfl_xor(m, off, 32));
    __shared__ float red[8];
    if ((threadIdx.x & 31) == 0) red[threadIdx.x >> 5] = m;
    __syncthreads();
    if (threadIdx.x == 0) {
        float r = red[0];
        for (int w = 1; w < 8; ++w) r = fmaxf(r, red[w]);
        smax[h] = r;
    }
}

// ---------------- fused masked softmax attention + P@V (flash style) --------
// block = 8 waves, 16 rows. nheads==8: wave -> head (64 cols). nheads==1:
// single head, wave -> 64-col slice of the 512-wide output. No online max:
// m_i = lrelu(s1_i + max_j s2_j) upper-bounds every score; masked entries
// exp to exactly 0. Per chunk: issue all 8 V-tile loads, then do the exp/mask
// VALU work (covers load latency), then 8 WMMAs.

__global__ __launch_bounds__(256)
void k_attn(const float* __restrict__ s1, const float* __restrict__ s2,
            const float* __restrict__ smax, const unsigned* __restrict__ mask,
            const __bf16* __restrict__ Vt, int nheads,
            __bf16* __restrict__ outb, float* __restrict__ outf) {
    int wave = threadIdx.x >> 5, lane = threadIdx.x & 31;
    int r0 = blockIdx.x * 16;
    int head = (nheads == 8) ? wave : 0;
    int col0 = (nheads == 8) ? head * 64 : wave * 64;
    int ln = lane & 15;
    int hi8 = (lane < 16) ? 0 : 8;
    int hi16 = (lane < 16) ? 0 : 16;
    int i = r0 + ln;

    const float* s2h = s2 + head * NN;
    float s1v = s1[head * NN + i];
    float mrow;
    {
        float t = s1v + smax[head];
        mrow = (t > 0.0f) ? t : ALPHA * t;
    }
    const unsigned* mrowp = mask + i * 128;

    v8f acc[4];
#pragma unroll
    for (int t = 0; t < 4; ++t)
#pragma unroll
        for (int e = 0; e < 8; ++e) acc[t][e] = 0.0f;
    float lsum = 0.0f;

    for (int jc = 0; jc < NN; jc += 64) {
        // -- issue all B-tile loads for this chunk first -------------------
        v16bf bv[8];
#pragma unroll
        for (int ks = 0; ks < 2; ++ks)
#pragma unroll
            for (int nt = 0; nt < 4; ++nt) {
                int vr = col0 + nt * 16 + ln;
                bv[ks * 4 + nt] =
                    *(const v16bf*)(Vt + vr * NN + jc + ks * 32 + hi16);
            }

        // prefetch next chunk's score block + mask words (global_prefetch_b8)
        int jn = (jc + 64) & (NN - 1);
        __builtin_prefetch(s2h + jn + hi8, 0, 1);
        __builtin_prefetch(mrowp + (jn >> 5), 0, 1);

        unsigned w0 = mrowp[(jc >> 5)];
        unsigned w1 = mrowp[(jc >> 5) + 1];

        // -- exp/mask/pack (overlaps the loads above) ----------------------
        v16bf a0, a1;
        #pragma unroll
        for (int g = 0; g < 4; ++g) {
            int jb = jc + (g >> 1) * 32 + hi16 + (g & 1) * 8;
            v4f lo = *(const v4f*)(s2h + jb);
            v4f hi = *(const v4f*)(s2h + jb + 4);
            unsigned wsel = (g < 2) ? w0 : w1;
#pragma unroll
            for (int t = 0; t < 8; ++t) {
                float sv = (t < 4) ? lo[t] : hi[t - 4];
                float e = s1v + sv;
                e = (e > 0.0f) ? e : ALPHA * e;
                int bit = (hi16 + (g & 1) * 8 + t) & 31;
                if (!((wsel >> bit) & 1u)) e = NEGV;
                float pe = __expf(e - mrow);
                lsum += pe;
                if (g < 2) a0[g * 8 + t] = (__bf16)pe;
                else       a1[(g - 2) * 8 + t] = (__bf16)pe;
            }
        }

        // -- 8 WMMAs -------------------------------------------------------
#pragma unroll
        for (int nt = 0; nt < 4; ++nt)
            acc[nt] = __builtin_amdgcn_wmma_f32_16x16x32_bf16(
                false, a0, false, bv[nt], (short)0, acc[nt], false, false);
#pragma unroll
        for (int nt = 0; nt < 4; ++nt)
            acc[nt] = __builtin_amdgcn_wmma_f32_16x16x32_bf16(
                false, a1, false, bv[4 + nt], (short)0, acc[nt], false, false);
    }

    float Lt = lsum + __shfl_xor(lsum, 16, 32);
    float Linv[8];
#pragma unroll
    for (int v = 0; v < 8; ++v) Linv[v] = 1.0f / __shfl(Lt, v + hi8, 32);

#pragma unroll
    for (int nt = 0; nt < 4; ++nt) {
        int col = col0 + nt * 16 + ln;
#pragma unroll
        for (int v = 0; v < 8; ++v) {
            int row = r0 + v + hi8;
            float o = acc[nt][v] * Linv[v];
            o = (o > 0.0f) ? o : (__expf(o) - 1.0f);   // ELU
            if (nheads == 8) outb[row * FD + col] = (__bf16)o;
            else             outf[row * FD + col] = o;
        }
    }
}

// ---------------- host orchestration ---------------------------------------

extern "C" void kernel_launch(void* const* d_in, const int* in_sizes, int n_in,
                              void* d_out, int out_size, void* d_ws, size_t ws_size,
                              hipStream_t stream) {
    const float* x   = (const float*)d_in[0];
    // d_in[1] = left (unused by the forward pass, matches reference)
    const float* adj = (const float*)d_in[2];
    const float* Ws  = (const float*)d_in[3];
    const float* a1  = (const float*)d_in[4];
    const float* a2  = (const float*)d_in[5];
    const float* Wo  = (const float*)d_in[6];
    const float* ao1 = (const float*)d_in[7];
    const float* ao2 = (const float*)d_in[8];

    char* ws = (char*)d_ws;
    __bf16*   xb    = (__bf16*)(ws + 0);                              // 4 MB
    __bf16*   WcatT = (__bf16*)(ws + (4ull  << 20));                  // 0.5 MB
    __bf16*   WoT   = (__bf16*)(ws + (4ull  << 20) + FD * FD * 2);    // 0.5 MB
    unsigned* mask  = (unsigned*)(ws + (5ull  << 20));                // 2 MB
    __bf16*   V     = (__bf16*)(ws + (7ull  << 20));                  // 4 MB (Wh, later Whc)
    __bf16*   Vt    = (__bf16*)(ws + (11ull << 20));                  // 4 MB
    __bf16*   hcatb = (__bf16*)(ws + (15ull << 20));                  // 4 MB
    float*    s1    = (float*)(ws + (19ull << 20));                   // 128 KB
    float*    s2    = (float*)(ws + (19ull << 20) + HEADS * NN * 4);  // 128 KB
    float*    smax  = (float*)(ws + (19ull << 20) + 2 * HEADS * NN * 4);

    dim3 b256(256);

    // phase 0: conversions / packing
    k_cvt_x  <<<dim3((NN * FD) / 256), b256, 0, stream>>>(x, xb);
    k_prep_w <<<dim3((FD * FD) / 256), b256, 0, stream>>>(Ws, Wo, WcatT, WoT);
    k_pack_adj<<<dim3((NN * NN) / 256), b256, 0, stream>>>(adj, mask);

    // phase 1: Wh = x @ Wcat  (all heads fused, bf16 WMMA)
    k_gemm<<<dim3(NN / 128, FD / 64), b256, 0, stream>>>(xb, WcatT, V);
    k_transpose<<<dim3(FD / 32, NN / 32), dim3(32, 8), 0, stream>>>(V, Vt);

    // phase 2: per-head scores + softmax bound
    k_scores<<<dim3((HEADS * NN) / 256), b256, 0, stream>>>(V, a1, a2, HEADS, NHD, s1, s2);
    k_smax<<<dim3(HEADS), b256, 0, stream>>>(s2, smax);

    // phase 3: fused masked attention per head, ELU -> hcat (bf16)
    k_attn<<<dim3(NN / 16), b256, 0, stream>>>(s1, s2, smax, mask, Vt, HEADS, hcatb, nullptr);

    // phase 4: Whc = hcat @ Wo  (reuse V/Vt)
    k_gemm<<<dim3(NN / 128, FD / 64), b256, 0, stream>>>(hcatb, WoT, V);
    k_transpose<<<dim3(FD / 32, NN / 32), dim3(32, 8), 0, stream>>>(V, Vt);

    // phase 5: output-layer scores + fused attention, ELU -> d_out (f32)
    k_scores<<<dim3(NN / 256), b256, 0, stream>>>(V, ao1, ao2, 1, FD, s1, s2);
    k_smax<<<dim3(1), b256, 0, stream>>>(s2, smax);
    k_attn<<<dim3(NN / 16), b256, 0, stream>>>(s1, s2, smax, mask, Vt, 1, nullptr,
                                               (float*)d_out);

    (void)in_sizes; (void)n_in; (void)out_size; (void)ws_size;
}
